// EnhancedGRUMemoryUpdater_26963804684871
// MI455X (gfx1250) — matrix operand order repaired and checked by reading, not compiled, not run on hardware
//
#include <hip/hip_runtime.h>

// ---------- types ----------
typedef __attribute__((ext_vector_type(16))) __bf16 v16bf;
typedef __attribute__((ext_vector_type(8)))  float  v8f;
typedef __attribute__((ext_vector_type(4)))  float  f4;

#define DEV __device__ __forceinline__

DEV v8f wmma_bf16(v16bf a, v16bf b, v8f c) {
  // D = A(16x32 bf16) * B(32x16 bf16) + C(16x16 f32)
  return __builtin_amdgcn_wmma_f32_16x16x32_bf16(
      /*neg_a=*/false, a, /*neg_b=*/false, b,
      /*c_mod=*/(short)0, c, /*reuse_a=*/false, /*reuse_b=*/false);
}

DEV v8f splat8(float x) {
  v8f r;
#pragma unroll
  for (int i = 0; i < 8; ++i) r[i] = x;
  return r;
}

// ---------- problem constants (from reference) ----------
constexpr int D     = 256;   // MEM_DIM
constexpr int MSG   = 172;   // MSG_DIM
constexpr int MSGP  = 192;   // padded to 6 k-tiles of 32
constexpr int FC    = 64;    // FC_DIM
constexpr int ROWS  = 32;    // rows (U) per block = 2 M-tiles of 16

// ---------- workspace layout: bf16 hi/lo B-fragments ----------
// fragment = 32 lanes x 16 bf16 = 512 bf16 = 1024 B; frag index = nt*ktiles + kt
constexpr size_t SZ_WIH   = (size_t)288 * 512 * 2;  // 48 ntiles * 6 ktiles
constexpr size_t SZ_WHH   = (size_t)384 * 512 * 2;  // 48 * 8
constexpr size_t SZ_WFC   = (size_t)32  * 512 * 2;  // 4 * 8
constexpr size_t SZ_WOUT  = (size_t)32  * 512 * 2;  // 16 * 2
constexpr size_t OFF_WIH_HI  = 0;
constexpr size_t OFF_WIH_LO  = OFF_WIH_HI  + SZ_WIH;
constexpr size_t OFF_WHH_HI  = OFF_WIH_LO  + SZ_WIH;
constexpr size_t OFF_WHH_LO  = OFF_WHH_HI  + SZ_WHH;
constexpr size_t OFF_WFC_HI  = OFF_WHH_LO  + SZ_WHH;
constexpr size_t OFF_WFC_LO  = OFF_WFC_HI  + SZ_WFC;
constexpr size_t OFF_WOUT_HI = OFF_WFC_LO  + SZ_WFC;
constexpr size_t OFF_WOUT_LO = OFF_WOUT_HI + SZ_WOUT;

// ---------- LDS layout ----------
constexpr int L_IDS  = 0;                       // 32 ints (pad to 256 B)
constexpr int L_X    = 256;                     // 32*192 f32 = 24576
constexpr int L_H    = L_X + ROWS * MSGP * 4;   // 32*256 f32 = 32768
constexpr int L_HN   = L_H + ROWS * D * 4;      // 32*256 f32 = 32768
constexpr int L_PR   = L_HN + ROWS * D * 4;     // 32*64  f32 = 8192
constexpr int L_FRAG = L_PR + ROWS * FC * 4;    // 56 slots * 1024 B = 57344
constexpr int SMEM_BYTES = L_FRAG + 56 * 1024;  // 155904 B

// =====================================================================
// Weight fragment preparation: W [N,K] row-major -> B-frag (W^T tiles)
// B layout: lane l <-> column nt*16 + l%16; bf16 j <-> K kt*32 + 8*(l/16) + (j&7) + 16*(j>>3)
// hi = RNE bf16(w);  lo = bf16(w - f32(hi))   (bf16x3 split)
// =====================================================================
__global__ void prep_frags(const float* __restrict__ W, int Kreal, int ktiles,
                           int nfrags, char* __restrict__ dhi, char* __restrict__ dlo) {
  int t = blockIdx.x * blockDim.x + threadIdx.x;
  if (t >= nfrags * 32) return;
  int l = t & 31, f = t >> 5;
  int kt = f % ktiles, nt = f / ktiles;
  int n = nt * 16 + (l & 15);
  v16bf vh, vl;
#pragma unroll
  for (int j = 0; j < 16; ++j) {
    int k = kt * 32 + ((l >> 4) << 3) + (j & 7) + ((j >> 3) << 4);
    float v = (k < Kreal) ? W[(size_t)n * Kreal + k] : 0.0f;
    __bf16 bh = (__bf16)v;
    vh[j] = bh;
    vl[j] = (__bf16)(v - (float)bh);
  }
  ((v16bf*)dhi)[f * 32 + l] = vh;
  ((v16bf*)dlo)[f * 32 + l] = vl;
}

// =====================================================================
// Streaming copy (nontemporal: keep L2 for weights/activations)
// =====================================================================
__global__ void copy_f4(const f4* __restrict__ src, f4* __restrict__ dst, long n) {
  long i = (long)blockIdx.x * blockDim.x + threadIdx.x;
  if (i < n) {
    f4 v = __builtin_nontemporal_load(src + i);
    __builtin_nontemporal_store(v, dst + i);
  }
}

// =====================================================================
// A-fragment builder: f32 LDS matrix -> bf16 hi/lo WMMA A operand in LDS
// A 16x32 bf16 layout: lane = M + 16*(K-half); j=0..7 -> K kb+{0..7},
// j=8..15 -> K kb+16+{0..7}; kb = kt*32 + 8*(lane>=16)
// =====================================================================
DEV void buildFrag(const float* src, int stride, int row, int kb,
                   char* fragBase, int slotHi, int slotLo, int lane) {
  v16bf vh, vl;
#pragma unroll
  for (int j = 0; j < 16; ++j) {
    int k = kb + (j & 7) + ((j >> 3) << 4);
    float v = src[row * stride + k];
    __bf16 bh = (__bf16)v;
    vh[j] = bh;
    vl[j] = (__bf16)(v - (float)bh);
  }
  *(v16bf*)(fragBase + slotHi * 1024 + lane * 32) = vh;
  *(v16bf*)(fragBase + slotLo * 1024 + lane * 32) = vl;
}

DEV v8f mma3(v16bf ah, v16bf al, v16bf bh, v16bf bl, v8f acc) {
  acc = wmma_bf16(ah, bh, acc);   // hi*hi
  acc = wmma_bf16(ah, bl, acc);   // hi*lo
  acc = wmma_bf16(al, bh, acc);   // lo*hi
  return acc;
}

// =====================================================================
// Fused GRU memory updater: 32 rows per block, 8 wave32 per block.
// =====================================================================
__global__ __launch_bounds__(256)
void gru_fused(const float* __restrict__ table,
               const int* __restrict__ ids,
               const float* __restrict__ msgs,
               const float* __restrict__ ts,
               const float* __restrict__ b_ih,
               const float* __restrict__ b_hh,
               const float* __restrict__ b_fc,
               const float* __restrict__ b_out,
               const char* __restrict__ wsb,
               float* __restrict__ outTab,
               float* __restrict__ outLU) {
  extern __shared__ __align__(32) char smem[];
  int*   idsL = (int*)(smem + L_IDS);
  float* xs   = (float*)(smem + L_X);
  float* hs   = (float*)(smem + L_H);
  float* hn   = (float*)(smem + L_HN);
  float* pr   = (float*)(smem + L_PR);
  char*  frag = smem + L_FRAG;

  const v16bf* WIHh  = (const v16bf*)(wsb + OFF_WIH_HI);
  const v16bf* WIHl  = (const v16bf*)(wsb + OFF_WIH_LO);
  const v16bf* WHHh  = (const v16bf*)(wsb + OFF_WHH_HI);
  const v16bf* WHHl  = (const v16bf*)(wsb + OFF_WHH_LO);
  const v16bf* WFCh  = (const v16bf*)(wsb + OFF_WFC_HI);
  const v16bf* WFCl  = (const v16bf*)(wsb + OFF_WFC_LO);
  const v16bf* WOUTh = (const v16bf*)(wsb + OFF_WOUT_HI);
  const v16bf* WOUTl = (const v16bf*)(wsb + OFF_WOUT_LO);

  const int tid  = threadIdx.x;
  const int w    = tid >> 5;     // wave id 0..7
  const int lane = tid & 31;
  const int lm   = lane & 15;
  const int kb8  = (lane >> 4) << 3;   // 0 or 8 (K-half select)
  const int base = blockIdx.x * ROWS;

  // --- ids + timestamps scatter ---
  if (tid < ROWS) {
    int id = ids[base + tid];
    idsL[tid] = id;
    outLU[id] = ts[base + tid];
  }
  __syncthreads();

  // --- stage x (zero-padded to 192) and gathered h into LDS ---
  for (int idx = tid; idx < ROWS * MSGP; idx += 256) {
    int row = idx / MSGP;
    int k   = idx - row * MSGP;
    xs[idx] = (k < MSG) ? msgs[(size_t)(base + row) * MSG + k] : 0.0f;
  }
  {
    const f4* t4 = (const f4*)table;
    f4* h4 = (f4*)hs;
    for (int idx = tid; idx < ROWS * (D / 4); idx += 256) {
      int row = idx >> 6, k4 = idx & 63;
      h4[idx] = t4[(size_t)idsL[row] * (D / 4) + k4];
    }
  }
  __syncthreads();

  // --- build A fragments for GEMM1: x (6 kt x 2 mt) slots 0..23, h (8x2) slots 24..55 ---
  for (int t = tid; t < 28 * 32; t += 256) {
    int l = t & 31, f = t >> 5;
    int lrow = l & 15, lkb = (l >> 4) << 3;
    if (f < 12) {
      int kt = f >> 1, mt = f & 1;
      buildFrag(xs, MSGP, mt * 16 + lrow, kt * 32 + lkb, frag, f * 2, f * 2 + 1, l);
    } else {
      int g = f - 12, kt = g >> 1, mt = g & 1;
      buildFrag(hs, D, mt * 16 + lrow, kt * 32 + lkb, frag, 24 + g * 2, 24 + g * 2 + 1, l);
    }
  }
  __syncthreads();

  // --- GEMM1: gates = [x|h] @ [W_ih^T; W_hh^T] + biases; gi_n / gh_n kept separate ---
  for (int ct = 0; ct < 2; ++ct) {
    const int ntr = 2 * w + ct, ntz = 16 + ntr, ntn = 32 + ntr;
    const int cr = ntr * 16 + lm, cz = ntz * 16 + lm, cn = ntn * 16 + lm;
    v8f aR[2], aZ[2], aGi[2], aGh[2];
    {
      float bR  = b_ih[cr] + b_hh[cr];
      float bZ  = b_ih[cz] + b_hh[cz];
      float bGi = b_ih[cn];
      float bGh = b_hh[cn];
#pragma unroll
      for (int mt = 0; mt < 2; ++mt) {
        aR[mt] = splat8(bR); aZ[mt] = splat8(bZ);
        aGi[mt] = splat8(bGi); aGh[mt] = splat8(bGh);
      }
    }
    // x @ W_ih^T  (K = 192 padded)
    for (int kt = 0; kt < 6; ++kt) {
      v16bf bhR = WIHh[(ntr * 6 + kt) * 32 + lane], blR = WIHl[(ntr * 6 + kt) * 32 + lane];
      v16bf bhZ = WIHh[(ntz * 6 + kt) * 32 + lane], blZ = WIHl[(ntz * 6 + kt) * 32 + lane];
      v16bf bhN = WIHh[(ntn * 6 + kt) * 32 + lane], blN = WIHl[(ntn * 6 + kt) * 32 + lane];
#pragma unroll
      for (int mt = 0; mt < 2; ++mt) {
        int f = (kt * 2 + mt) * 2;
        v16bf ah = *(const v16bf*)(frag + f * 1024 + lane * 32);
        v16bf al = *(const v16bf*)(frag + (f + 1) * 1024 + lane * 32);
        aR[mt]  = mma3(ah, al, bhR, blR, aR[mt]);
        aZ[mt]  = mma3(ah, al, bhZ, blZ, aZ[mt]);
        aGi[mt] = mma3(ah, al, bhN, blN, aGi[mt]);
      }
    }
    // h @ W_hh^T  (K = 256)
    for (int kt = 0; kt < 8; ++kt) {
      v16bf bhR = WHHh[(ntr * 8 + kt) * 32 + lane], blR = WHHl[(ntr * 8 + kt) * 32 + lane];
      v16bf bhZ = WHHh[(ntz * 8 + kt) * 32 + lane], blZ = WHHl[(ntz * 8 + kt) * 32 + lane];
      v16bf bhN = WHHh[(ntn * 8 + kt) * 32 + lane], blN = WHHl[(ntn * 8 + kt) * 32 + lane];
#pragma unroll
      for (int mt = 0; mt < 2; ++mt) {
        int f = 24 + (kt * 2 + mt) * 2;
        v16bf ah = *(const v16bf*)(frag + f * 1024 + lane * 32);
        v16bf al = *(const v16bf*)(frag + (f + 1) * 1024 + lane * 32);
        aR[mt]  = mma3(ah, al, bhR, blR, aR[mt]);
        aZ[mt]  = mma3(ah, al, bhZ, blZ, aZ[mt]);
        aGh[mt] = mma3(ah, al, bhN, blN, aGh[mt]);
      }
    }
    // GRU elementwise -> h_new (LDS)
#pragma unroll
    for (int mt = 0; mt < 2; ++mt) {
#pragma unroll
      for (int i = 0; i < 8; ++i) {
        int rl = mt * 16 + i + kb8;
        float r  = 1.0f / (1.0f + __expf(-aR[mt][i]));
        float z  = 1.0f / (1.0f + __expf(-aZ[mt][i]));
        float nn = tanhf(aGi[mt][i] + r * aGh[mt][i]);
        float hv = hs[rl * D + cr];
        hn[rl * D + cr] = (1.0f - z) * nn + z * hv;
      }
    }
  }
  __syncthreads();

  // --- A fragments of h_new (8 kt x 2 mt -> slots 0..31) ---
  for (int t = tid; t < 16 * 32; t += 256) {
    int l = t & 31, f = t >> 5;
    int kt = f >> 1, mt = f & 1;
    buildFrag(hn, D, mt * 16 + (l & 15), kt * 32 + ((l >> 4) << 3), frag, f * 2, f * 2 + 1, l);
  }
  __syncthreads();

  // --- GEMM3: pred = h_new @ W_fc^T + b_fc  (N=64, K=256) ---
  {
    int nt = w & 3, mt = w >> 2;
    int c = nt * 16 + lm;
    v8f acc = splat8(b_fc[c]);
    for (int kt = 0; kt < 8; ++kt) {
      v16bf bh = WFCh[(nt * 8 + kt) * 32 + lane], bl = WFCl[(nt * 8 + kt) * 32 + lane];
      int f = (kt * 2 + mt) * 2;
      v16bf ah = *(const v16bf*)(frag + f * 1024 + lane * 32);
      v16bf al = *(const v16bf*)(frag + (f + 1) * 1024 + lane * 32);
      acc = mma3(ah, al, bh, bl, acc);
    }
#pragma unroll
    for (int i = 0; i < 8; ++i) {
      int rl = mt * 16 + i + kb8;
      pr[rl * FC + c] = acc[i];
    }
  }
  __syncthreads();

  // --- A fragments of pred (2 kt x 2 mt -> slots 0..7) ---
  if (tid < 4 * 32) {
    int l = tid & 31, f = tid >> 5;
    int kt = f >> 1, mt = f & 1;
    buildFrag(pr, FC, mt * 16 + (l & 15), kt * 32 + ((l >> 4) << 3), frag, f * 2, f * 2 + 1, l);
  }
  __syncthreads();

  // --- GEMM4: out = pred @ W_out^T + b_out (N=256, K=64) -> scatter by id ---
#pragma unroll
  for (int mt = 0; mt < 2; ++mt) {
#pragma unroll
    for (int nh = 0; nh < 2; ++nh) {
      int nt = w * 2 + nh;
      int c = nt * 16 + lm;
      v8f acc = splat8(b_out[c]);
#pragma unroll
      for (int kt = 0; kt < 2; ++kt) {
        v16bf bh = WOUTh[(nt * 2 + kt) * 32 + lane], bl = WOUTl[(nt * 2 + kt) * 32 + lane];
        int f = (kt * 2 + mt) * 2;
        v16bf ah = *(const v16bf*)(frag + f * 1024 + lane * 32);
        v16bf al = *(const v16bf*)(frag + (f + 1) * 1024 + lane * 32);
        acc = mma3(ah, al, bh, bl, acc);
      }
#pragma unroll
      for (int i = 0; i < 8; ++i) {
        int rl = mt * 16 + i + kb8;
        outTab[(size_t)idsL[rl] * D + c] = acc[i];
      }
    }
  }
}

// =====================================================================
extern "C" void kernel_launch(void* const* d_in, const int* in_sizes, int n_in,
                              void* d_out, int out_size, void* d_ws, size_t ws_size,
                              hipStream_t stream) {
  (void)n_in; (void)out_size; (void)ws_size;
  const float* table = (const float*)d_in[0];
  const float* lastu = (const float*)d_in[1];
  const int*   ids   = (const int*)d_in[2];
  const float* msgs  = (const float*)d_in[3];
  const float* ts    = (const float*)d_in[4];
  const float* W_ih  = (const float*)d_in[5];
  const float* W_hh  = (const float*)d_in[6];
  const float* b_ih  = (const float*)d_in[7];
  const float* b_hh  = (const float*)d_in[8];
  const float* W_fc  = (const float*)d_in[9];
  const float* b_fc  = (const float*)d_in[10];
  const float* W_out = (const float*)d_in[11];
  const float* b_out = (const float*)d_in[12];

  const int  U       = in_sizes[2];            // 131072
  const long nNodes  = in_sizes[1];            // 500000
  const long tabElem = nNodes * (long)D;

  float* outTab = (float*)d_out;
  float* outLU  = outTab + tabElem;
  char*  ws     = (char*)d_ws;

  // 1) weight -> bf16 hi/lo B-fragments (L2-resident for the whole GEMM phase)
  prep_frags<<<(288 * 32 + 255) / 256, 256, 0, stream>>>(W_ih, 172, 6, 288,
      ws + OFF_WIH_HI, ws + OFF_WIH_LO);
  prep_frags<<<(384 * 32 + 255) / 256, 256, 0, stream>>>(W_hh, 256, 8, 384,
      ws + OFF_WHH_HI, ws + OFF_WHH_LO);
  prep_frags<<<(32 * 32 + 255) / 256, 256, 0, stream>>>(W_fc, 256, 8, 32,
      ws + OFF_WFC_HI, ws + OFF_WFC_LO);
  prep_frags<<<(32 * 32 + 255) / 256, 256, 0, stream>>>(W_out, 64, 2, 32,
      ws + OFF_WOUT_HI, ws + OFF_WOUT_LO);

  // 2) streaming copy of the base table + last_update into the output
  long n4t = tabElem / 4;
  copy_f4<<<(int)((n4t + 255) / 256), 256, 0, stream>>>((const f4*)table, (f4*)outTab, n4t);
  long n4l = nNodes / 4;
  copy_f4<<<(int)((n4l + 255) / 256), 256, 0, stream>>>((const f4*)lastu, (f4*)outLU, n4l);

  // 3) fused GRU update + scatter (U assumed multiple of 32; 131072/32 = 4096)
  gru_fused<<<U / ROWS, 256, SMEM_BYTES, stream>>>(
      table, ids, msgs, ts, b_ih, b_hh, b_fc, b_out, ws, outTab, outLU);
}
